// CodebookLayer_52115133170014
// MI455X (gfx1250) — matrix operand order repaired and checked by reading, not compiled, not run on hardware
//
#include <hip/hip_runtime.h>

typedef __bf16 bf16_t;
typedef __attribute__((ext_vector_type(16))) __bf16 v16bf;
typedef __attribute__((ext_vector_type(8)))  __bf16 v8bf;
typedef __attribute__((ext_vector_type(8)))  float  v8f;
typedef __attribute__((ext_vector_type(4)))  unsigned int u32x4;
typedef __attribute__((ext_vector_type(4)))  int i32x4;
typedef __attribute__((ext_vector_type(8)))  int i32x8;

#define NTOK   8192        // B*S
#define DDIM   1024
#define NCODE  16384
#define KSEL   8
#define TILE_M 32
#define NWAVES 8
#define NTILES (NCODE / 16)   // 1024 code tiles of 16

// float -> bf16 round-to-nearest-even without relying on cast codegen
__device__ __forceinline__ bf16_t f2bf(float f) {
  unsigned u = __float_as_uint(f);
  unsigned r = (u + 0x7FFFu + ((u >> 16) & 1u)) >> 16;
  unsigned short h = (unsigned short)r;
  bf16_t b;
  __builtin_memcpy(&b, &h, 2);
  return b;
}

__global__ __launch_bounds__(256) void cvt_inputs_kernel(const float* __restrict__ x,
                                                         bf16_t* __restrict__ xb) {
  size_t i = ((size_t)blockIdx.x * 256 + threadIdx.x) * 4;
  const float4 v = *(const float4*)(x + i);
  bf16_t o[4] = {f2bf(v.x), f2bf(v.y), f2bf(v.z), f2bf(v.w)};
  *(uint2*)(xb + i) = *(const uint2*)o;
}

__global__ __launch_bounds__(256) void cvt_codebook_kernel(const float* __restrict__ cb,
                                                           bf16_t* __restrict__ cbb,
                                                           float* __restrict__ c2) {
  const int c   = blockIdx.x;
  const int tid = threadIdx.x;
  const size_t base = (size_t)c * DDIM + (size_t)tid * 4;
  const float4 v = *(const float4*)(cb + base);
  bf16_t o[4] = {f2bf(v.x), f2bf(v.y), f2bf(v.z), f2bf(v.w)};
  *(uint2*)(cbb + base) = *(const uint2*)o;

  __shared__ float red[256];
  red[tid] = v.x * v.x + v.y * v.y + v.z * v.z + v.w * v.w;
  __syncthreads();
  for (int s = 128; s > 0; s >>= 1) {
    if (tid < s) red[tid] += red[tid + s];
    __syncthreads();
  }
  if (tid == 0) c2[c] = red[0];
}

// branch-free sorted-descending insertion into an 8-entry list
__device__ __forceinline__ void topk_insert(float val, int idx,
                                            float (&bv)[KSEL], int (&bi)[KSEL]) {
#pragma unroll
  for (int s = 0; s < KSEL; ++s) {
    bool gt = val > bv[s];
    float tv = bv[s]; int ti = bi[s];
    bv[s] = gt ? val : tv;
    bi[s] = gt ? idx : ti;
    val   = gt ? tv  : val;
    idx   = gt ? ti  : idx;
  }
}

__global__ __launch_bounds__(256) void codebook_topk_kernel(
    const bf16_t* __restrict__ xb, const bf16_t* __restrict__ cbb,
    const float* __restrict__ c2, const float* __restrict__ cbf,
    float* __restrict__ out) {
  __shared__ __align__(16) bf16_t ldsA[TILE_M * DDIM];        // 64 KB A tile
  __shared__ float  ldsLog[NWAVES][TILE_M * 17];              // padded logit transpose
  __shared__ float  mergeV[NWAVES][TILE_M][KSEL];
  __shared__ int    mergeI[NWAVES][TILE_M][KSEL];
  __shared__ int    finalI[TILE_M][KSEL];

  const int tid  = threadIdx.x;
  const int lane = tid & 31;
  const int wave = tid >> 5;
  const int tok0 = blockIdx.x * TILE_M;

  // ---- stage 0: stage A tile (32 tokens x 1024 bf16 = 64 KB) into LDS ----
#if __has_builtin(__builtin_amdgcn_tensor_load_to_lds)
  if (wave == 0) {
    // 1-D TDM descriptor: 8192 x 8-byte elements, contiguous.
    unsigned ldsOff =
        (unsigned)(unsigned long long)(__attribute__((address_space(3))) bf16_t*)ldsA;
    unsigned long long ga = (unsigned long long)(xb + (size_t)tok0 * DDIM);
    u32x4 g0;
    g0[0] = 1u;                                              // count=1, user D#
    g0[1] = ldsOff;                                          // lds_addr
    g0[2] = (unsigned)(ga & 0xFFFFFFFFu);                    // global_addr[31:0]
    g0[3] = (unsigned)((ga >> 32) & 0x01FFFFFFu) | (2u << 30); // addr[56:32] | type=2
    i32x8 g1;
    g1[0] = 0x00030000;          // workgroup_mask=0, data_size=3 (8B)
    g1[1] = 0x20000000;          // tensor_dim0 = 8192 (low16 in bits 63:48)
    g1[2] = 0x00010000;          // tensor_dim0 hi=0, tensor_dim1 = 1
    g1[3] = 0x20000000;          // tensor_dim1 hi=0, tile_dim0 = 8192
    g1[4] = 1;                   // tile_dim1 = 1, tile_dim2 = 0
    g1[5] = 8192;                // tensor_dim0_stride[31:0]
    g1[6] = 0;
    g1[7] = 0;
    i32x4 z4 = {0, 0, 0, 0};
#if __clang_major__ >= 23
    i32x8 z8 = {0, 0, 0, 0, 0, 0, 0, 0};
    __builtin_amdgcn_tensor_load_to_lds(g0, g1, z4, z4, z8, 0);
#else
    __builtin_amdgcn_tensor_load_to_lds(g0, g1, z4, z4, 0);
#endif
#if __has_builtin(__builtin_amdgcn_s_wait_tensorcnt)
    __builtin_amdgcn_s_wait_tensorcnt(0);
#else
    asm volatile("s_wait_tensorcnt 0x0" ::: "memory");
#endif
  }
#else
  {
    const uint4* src = (const uint4*)(xb + (size_t)tok0 * DDIM);
    uint4* dst = (uint4*)ldsA;
#pragma unroll
    for (int i = 0; i < 16; ++i) dst[tid * 16 + i] = src[tid * 16 + i];
  }
#endif
  __syncthreads();

  float bv[KSEL]; int bi[KSEL];
#pragma unroll
  for (int s = 0; s < KSEL; ++s) { bv[s] = -3.0e38f; bi[s] = 0; }

  const int half  = lane >> 4;      // 0 or 1 (wave halves per ISA layouts)
  const int l16   = lane & 15;
  const int aKoff = half * 8;       // A: lanes 0-15 hold K{0..7,16..23}; 16-31 hold K{8..15,24..31}
  const int bKoff = half * 8;       // B: lanes 0-15 hold K{0..7,16..23} of col n; 16-31 hold K{8..15,24..31}

  // ---- stage 1: fused WMMA distance GEMM + per-wave running top-8 ----
  // Two 16-row accumulators share every B fragment: 2 WMMAs per B load.
  for (int ct = wave; ct < NTILES; ct += NWAVES) {
    const int code0 = ct * 16;
    const bf16_t* brow = cbb + (size_t)(code0 + l16) * DDIM + bKoff;
    __builtin_prefetch(cbb + (size_t)(code0 + NWAVES * 16 + l16) * DDIM, 0, 1);

    v8f acc0 = {};
    v8f acc1 = {};
#pragma unroll 2
    for (int kk = 0; kk < DDIM; kk += 32) {
      const bf16_t* arow0 = ldsA + (size_t)l16 * DDIM + kk + aKoff;
      const bf16_t* arow1 = arow0 + 16 * DDIM;
      v8bf a0lo = *(const v8bf*)(arow0);
      v8bf a0hi = *(const v8bf*)(arow0 + 16);
      v8bf a1lo = *(const v8bf*)(arow1);
      v8bf a1hi = *(const v8bf*)(arow1 + 16);
      v16bf a0 = __builtin_shufflevector(a0lo, a0hi, 0, 1, 2, 3, 4, 5, 6, 7,
                                         8, 9, 10, 11, 12, 13, 14, 15);
      v16bf a1 = __builtin_shufflevector(a1lo, a1hi, 0, 1, 2, 3, 4, 5, 6, 7,
                                         8, 9, 10, 11, 12, 13, 14, 15);
      v8bf blo = *(const v8bf*)(brow + kk);
      v8bf bhi = *(const v8bf*)(brow + kk + 16);
      v16bf b = __builtin_shufflevector(blo, bhi, 0, 1, 2, 3, 4, 5, 6, 7,
                                        8, 9, 10, 11, 12, 13, 14, 15);
      acc0 = __builtin_amdgcn_wmma_f32_16x16x32_bf16(false, a0, false, b,
                                                     (short)0, acc0, false, false);
      acc1 = __builtin_amdgcn_wmma_f32_16x16x32_bf16(false, a1, false, b,
                                                     (short)0, acc1, false, false);
    }

    // score = 2*xc - c2 (x^2 is row-constant: ranking-invariant)
    const float c2v = c2[code0 + l16];
#pragma unroll
    for (int r = 0; r < 8; ++r) {
      const int m = r + half * 8;                    // C layout: vgpr r, half-wave -> row
      ldsLog[wave][m * 17 + l16]        = 2.0f * acc0[r] - c2v;
      ldsLog[wave][(16 + m) * 17 + l16] = 2.0f * acc1[r] - c2v;
    }
    asm volatile("s_wait_dscnt 0" ::: "memory");     // LDS writes visible to whole wave

    // all 32 lanes merge: lane = token within tile
#pragma unroll
    for (int j = 0; j < 16; ++j)
      topk_insert(ldsLog[wave][lane * 17 + j], code0 + j, bv, bi);
  }

#pragma unroll
  for (int s = 0; s < KSEL; ++s) {
    mergeV[wave][lane][s] = bv[s];
    mergeI[wave][lane][s] = bi[s];
  }
  __syncthreads();

  // ---- stage 2: cross-wave merge (64 candidates -> final top-8 per token) ----
  if (tid < TILE_M) {
    float fv[KSEL]; int fi[KSEL];
#pragma unroll
    for (int s = 0; s < KSEL; ++s) { fv[s] = -3.0e38f; fi[s] = 0; }
    for (int w = 0; w < NWAVES; ++w)
#pragma unroll
      for (int s = 0; s < KSEL; ++s)
        topk_insert(mergeV[w][tid][s], mergeI[w][tid][s], fv, fi);

    float* outIds = out + (size_t)NTOK * DDIM;       // ids flattened after outputs
#pragma unroll
    for (int s = 0; s < KSEL; ++s) {
      finalI[tid][s] = fi[s];
      outIds[(size_t)(tok0 + tid) * KSEL + s] = (float)fi[s];
    }
  }
  __syncthreads();

  // ---- stage 3: gather top-8 f32 codebook rows and average ----
  const int t  = tid >> 3;        // token within tile (8 threads per token)
  const int c8 = tid & 7;
  int cs[KSEL];
#pragma unroll
  for (int s = 0; s < KSEL; ++s) cs[s] = finalI[t][s];
  float* orow = out + (size_t)(tok0 + t) * DDIM;
  for (int d0 = c8 * 4; d0 < DDIM; d0 += 32) {
    float ax = 0.f, ay = 0.f, az = 0.f, aw = 0.f;
#pragma unroll
    for (int s = 0; s < KSEL; ++s) {
      const float4 v = *(const float4*)(cbf + (size_t)cs[s] * DDIM + d0);
      ax += v.x; ay += v.y; az += v.z; aw += v.w;
    }
    float4 o;
    o.x = ax * 0.125f; o.y = ay * 0.125f; o.z = az * 0.125f; o.w = aw * 0.125f;
    *(float4*)(orow + d0) = o;
  }
}

extern "C" void kernel_launch(void* const* d_in, const int* in_sizes, int n_in,
                              void* d_out, int out_size, void* d_ws, size_t ws_size,
                              hipStream_t stream) {
  const float* x  = (const float*)d_in[0];   // inputs   [B,S,D] f32
  const float* cb = (const float*)d_in[1];   // codebook [C,D]   f32
  // d_in[2] = kcodes (fixed at 8)

  bf16_t* xb  = (bf16_t*)d_ws;                                    // 16 MB
  bf16_t* cbb = (bf16_t*)((char*)d_ws + ((size_t)16 << 20));      // 32 MB
  float*  c2  = (float*)((char*)d_ws + ((size_t)48 << 20));       // 64 KB

  cvt_inputs_kernel<<<(NTOK * DDIM) / 1024, 256, 0, stream>>>(x, xb);
  cvt_codebook_kernel<<<NCODE, 256, 0, stream>>>(cb, cbb, c2);
  codebook_topk_kernel<<<NTOK / TILE_M, 256, 0, stream>>>(xb, cbb, c2, cb,
                                                          (float*)d_out);
}
